// SDGraphUNet_25383256720093
// MI455X (gfx1250) — matrix operand order repaired and checked by reading, not compiled, not run on hardware
//
#include <hip/hip_runtime.h>
#include <hip/hip_bf16.h>

// ---------------------------------------------------------------------------
// Shapes (fixed by reference setup_inputs)
// ---------------------------------------------------------------------------
#define BS     4
#define NSTK   32
#define NPNT   256
#define NPTS   8192          // NSTK * NPNT
#define SP_IN  64
#define DN_IN  32
#define CFEA   96            // SP_IN + DN_IN
#define TWO_C  192
#define MID_DN 110
#define MIDP   112           // MID_DN padded to 16
#define DOUT   64
#define MID_SP 156
#define SP_OUT 128
#define KDN    50
#define KSP    2
#define LEAK   0.2f
#define NEG_INF (-3.402823466e38f)

#define NKB1   6             // K tiles (of 32) in GEMM1 (192/32)
#define NKB2   4             // K tiles in GEMM2 (128/32, K padded 112->128)
#define NNT1   7             // N tiles in GEMM1 (112/16)
#define NNT2   4             // N tiles in GEMM2 (64/16)

typedef __bf16 bf16;
typedef __attribute__((ext_vector_type(16))) __bf16 v16bf;
typedef __attribute__((ext_vector_type(8)))  float  v8f;
typedef __attribute__((ext_vector_type(8)))  unsigned int u32x8;   // 32B chunk

__device__ __forceinline__ float lrelu(float v) { return v > 0.f ? v : LEAK * v; }

// ---------------------------------------------------------------------------
// Fragment swizzle helpers (ISA 7.12.2, wave32, 16-bit operands)
// A 16x32: lane = (M&15) + ahi*16, elem e; K = e + (e>=8?8:0) + ahi*8
//   inverse: ahi = (K>>3)&1, e = (K&7) + (K>=16?8:0)
// B 32x16: lane = (N&15) + bhi*16, elem e; K = (e&7) + bhi*8 + (e>>3)*16
//   inverse: bhi = (K>>3)&1, e = (K&7) + (K>=16?8:0)
// C/D 16x16 f32: lane L (N = L&15), vgpr r -> M = r + (L>=16 ? 8 : 0)
// Swizzled storage: frag-major [tile][kb][lane][e], 32B per lane-fragment.
// ---------------------------------------------------------------------------
__device__ __forceinline__ int a_swz_pos(int mtile, int nkb, int kb, int m, int kk) {
  const int ahi = (kk >> 3) & 1;
  const int e   = (kk & 7) + (kk >= 16 ? 8 : 0);
  return (((mtile * nkb + kb) * 32) + m + ahi * 16) * 16 + e;
}

// ---------------------------------------------------------------------------
// Kernel 1: dense point features xt[b][n][96]  (dense ch 0..31, sparse ch 32..95)
// plus x2[b][n] = ||x||^2
// ---------------------------------------------------------------------------
__global__ __launch_bounds__(256) void build_xt(const float* __restrict__ sparse,
                                                const float* __restrict__ dense,
                                                float* __restrict__ xtf,
                                                float* __restrict__ x2) {
  const int blk = blockIdx.x;               // b*NSTK + s
  const int b = blk / NSTK, s = blk % NSTK;
  const int p = threadIdx.x;
  const int n = s * NPNT + p;
  float sum = 0.f;
  float* dst = xtf + ((size_t)b * NPTS + n) * CFEA;
#pragma unroll 4
  for (int c = 0; c < DN_IN; ++c) {
    const float v = dense[(((size_t)b * DN_IN + c) * NSTK + s) * NPNT + p];
    dst[c] = v; sum += v * v;
  }
#pragma unroll 4
  for (int c = 0; c < SP_IN; ++c) {
    const float v = sparse[((size_t)b * SP_IN + c) * NSTK + s];
    dst[DN_IN + c] = v; sum += v * v;
  }
  x2[b * NPTS + n] = sum;
}

// ---------------------------------------------------------------------------
// Kernel 2: sparse node features spxt[b][32][96]
// ---------------------------------------------------------------------------
__global__ __launch_bounds__(96) void build_sp_xt(const float* __restrict__ sparse,
                                                  const float* __restrict__ dense,
                                                  float* __restrict__ spxt) {
  const int b = blockIdx.x;
  const int c = threadIdx.x;                // 0..95
  for (int s = 0; s < NSTK; ++s) {
    float v;
    if (c < SP_IN) {
      v = sparse[((size_t)b * SP_IN + c) * NSTK + s];
    } else {
      v = NEG_INF;
      const float* src = dense + (((size_t)b * DN_IN + (c - SP_IN)) * NSTK + s) * NPNT;
      for (int p = 0; p < NPNT; ++p) v = fmaxf(v, src[p]);
    }
    spxt[((size_t)b * NSTK + s) * CFEA + c] = v;
  }
}

// ---------------------------------------------------------------------------
// Kernel 3: dense MLP weights -> bf16, B-fragment swizzled layout.
// w1sw: [nt=7][kb=6][lane=32][e=16]  from dn_W1 [110][192]
// w2sw: [nt=4][kb=4][lane=32][e=16]  from dn_W2 [64][110] (K padded to 128)
// ---------------------------------------------------------------------------
__global__ __launch_bounds__(256) void conv_weights(const float* __restrict__ dnW1,
                                                    const float* __restrict__ dnW2,
                                                    bf16* __restrict__ w1sw,
                                                    bf16* __restrict__ w2sw) {
  const int t = blockIdx.x * 256 + threadIdx.x;
  if (t < NNT1 * NKB1 * 512) {
    const int e    = t & 15;
    const int lane = (t >> 4) & 31;
    const int kb   = (t >> 9) % NKB1;
    const int nt   = t / (512 * NKB1);
    const int n  = nt * 16 + (lane & 15);
    const int k  = kb * 32 + (e & 7) + (lane >> 4) * 8 + (e >> 3) * 16;
    w1sw[t] = (bf16)(n < MID_DN ? dnW1[n * TWO_C + k] : 0.f);
  }
  if (t < NNT2 * NKB2 * 512) {
    const int e    = t & 15;
    const int lane = (t >> 4) & 31;
    const int kb   = (t >> 9) & 3;
    const int nt   = t >> 11;
    const int n  = nt * 16 + (lane & 15);
    const int k  = kb * 32 + (e & 7) + (lane >> 4) * 8 + (e >> 3) * 16;
    w2sw[t] = (bf16)(k < MID_DN ? dnW2[n * MID_DN + k] : 0.f);
  }
}

// ---------------------------------------------------------------------------
// Kernel 4: dense KNN (k=50).  16 query rows per workgroup; stream j in tiles
// of 256; maintain descending-sorted top-50 (strict > keeps jax top_k order).
// ---------------------------------------------------------------------------
#define TI 16
__global__ __launch_bounds__(256) void knn_topk(const float* __restrict__ xtf,
                                                const float* __restrict__ x2,
                                                int* __restrict__ knn) {
  __shared__ __align__(16) float sxi[TI][CFEA];
  __shared__ float sx2i[TI];
  __shared__ float dtile[TI][256];
  __shared__ float tvals[TI][KDN];
  __shared__ int   tidxs[TI][KDN];

  const int blk = blockIdx.x;
  const int b  = blk / (NPTS / TI);
  const int i0 = (blk % (NPTS / TI)) * TI;
  const int tid = threadIdx.x;

  for (int t = tid; t < TI * CFEA; t += 256) {
    const int ti = t / CFEA, c = t % CFEA;
    sxi[ti][c] = xtf[((size_t)b * NPTS + i0 + ti) * CFEA + c];
  }
  if (tid < TI) sx2i[tid] = x2[b * NPTS + i0 + tid];
  for (int t = tid; t < TI * KDN; t += 256) {
    tvals[t / KDN][t % KDN] = NEG_INF;
    tidxs[t / KDN][t % KDN] = 0;
  }
  __syncthreads();

  for (int j0 = 0; j0 < NPTS; j0 += 256) {
    const int j = j0 + tid;
    const float4* xj4 = (const float4*)(xtf + ((size_t)b * NPTS + j) * CFEA);
    const float4 (*sxi4)[CFEA / 4] = (const float4 (*)[CFEA / 4])sxi;
    float acc[TI];
#pragma unroll
    for (int ti = 0; ti < TI; ++ti) acc[ti] = 0.f;
    for (int c4 = 0; c4 < CFEA / 4; ++c4) {
      const float4 q = xj4[c4];
#pragma unroll
      for (int ti = 0; ti < TI; ++ti) {
        const float4 s = sxi4[ti][c4];
        acc[ti] += s.x * q.x + s.y * q.y + s.z * q.z + s.w * q.w;
      }
    }
    const float x2j = x2[b * NPTS + j];
#pragma unroll
    for (int ti = 0; ti < TI; ++ti)
      dtile[ti][tid] = 2.f * acc[ti] - sx2i[ti] - x2j;
    __syncthreads();

    if (tid < TI) {
      const int ti = tid;
      float mn = tvals[ti][KDN - 1];
      for (int jj = 0; jj < 256; ++jj) {
        const float v = dtile[ti][jj];
        if (v > mn) {
          int p = KDN - 1;
          while (p > 0 && v > tvals[ti][p - 1]) {
            tvals[ti][p] = tvals[ti][p - 1];
            tidxs[ti][p] = tidxs[ti][p - 1];
            --p;
          }
          tvals[ti][p] = v;
          tidxs[ti][p] = j0 + jj;
          mn = tvals[ti][KDN - 1];
        }
      }
    }
    __syncthreads();
  }

  for (int t = tid; t < TI * KDN; t += 256)
    knn[((size_t)b * NPTS + i0 + t / KDN) * KDN + (t % KDN)] = tidxs[t / KDN][t % KDN];
}

// ---------------------------------------------------------------------------
// Kernel 5 (WMMA): per point, edge matrix (64x192, 50 real rows) staged in LDS
// in A-fragment swizzled bf16 layout; H1 = lrelu(E@W1+b1) restaged swizzled;
// H2 = lrelu(H1@W2+b2); out = max over 50 edge rows.
// 4 waves / block, wave w owns M-rows [16w, 16w+16).
// Fragment loads are single 32B vector loads (ds/global b128 pairs).
// ---------------------------------------------------------------------------
__global__ __launch_bounds__(128) void dense_edge_gemm(const float* __restrict__ xtf,
                                                       const int* __restrict__ knn,
                                                       const bf16* __restrict__ w1sw,
                                                       const float* __restrict__ b1,
                                                       const bf16* __restrict__ w2sw,
                                                       const float* __restrict__ b2,
                                                       float* __restrict__ dnout) {
  __shared__ u32x8 sEv[4 * NKB1 * 32];     // 24 KB, A-swizzled edge matrix
  __shared__ u32x8 sH1v[4 * NKB2 * 32];    // 16 KB, A-swizzled H1 (K pad zeroed)
  __shared__ float sH2[64][DOUT];          // 16 KB

  bf16* sE  = (bf16*)sEv;
  bf16* sH1 = (bf16*)sH1v;

  const int blk = blockIdx.x;
  const int b = blk / NPTS, n = blk % NPTS;
  const int tid = threadIdx.x;
  const float4* xi4 = (const float4*)(xtf + ((size_t)b * NPTS + n) * CFEA);
  const int*    nb  = knn + ((size_t)b * NPTS + n) * KDN;

  // Zero swizzled H1 (so the K=112..127 pad is zero).
  for (int t = tid; t < 4 * NKB2 * 32; t += 128) {
    u32x8 z = {0, 0, 0, 0, 0, 0, 0, 0};
    sH1v[t] = z;
  }

  // Build edge matrix, 4 consecutive K per thread-iteration (never straddles
  // an e/hi boundary -> 4 consecutive bf16 in the swizzled layout).
  for (int t = tid; t < 64 * (TWO_C / 4); t += 128) {
    const int r  = t / (TWO_C / 4);
    const int c4 = t % (TWO_C / 4);
    const int c  = c4 * 4;
    float vx = 0.f, vy = 0.f, vz = 0.f, vw = 0.f;
    if (r < KDN) {
      const int j = nb[r];
      const float4* xj4 = (const float4*)(xtf + ((size_t)b * NPTS + j) * CFEA);
      if (c4 < CFEA / 4) {
        const float4 a = xj4[c4], q = xi4[c4];
        vx = a.x - q.x; vy = a.y - q.y; vz = a.z - q.z; vw = a.w - q.w;
      } else {
        const float4 q = xi4[c4 - CFEA / 4];
        vx = q.x; vy = q.y; vz = q.z; vw = q.w;
      }
    }
    const int pos = a_swz_pos(r >> 4, NKB1, c >> 5, r & 15, c & 31);
    bf16* dst = sE + pos;
    dst[0] = (bf16)vx; dst[1] = (bf16)vy; dst[2] = (bf16)vz; dst[3] = (bf16)vw;
  }
  __syncthreads();

  const int wave = tid >> 5, lane = tid & 31;
  const int nlo = lane & 15, hi = (lane >> 4) & 1;
  const int rbase = wave * 16;

  // GEMM1: H1[64][112] = lrelu(E[64][192] @ W1[192][112] + b1)
  const u32x8* w1v = (const u32x8*)w1sw;
  for (int nt = 0; nt < NNT1; ++nt) {
    const int col = nt * 16 + nlo;
    const float bias = (col < MID_DN) ? b1[col] : 0.f;
    v8f acc;
#pragma unroll
    for (int r = 0; r < 8; ++r) acc[r] = bias;
#pragma unroll
    for (int kb = 0; kb < NKB1; ++kb) {
      v16bf a  = __builtin_bit_cast(v16bf, sEv[(wave * NKB1 + kb) * 32 + lane]);
      v16bf bm = __builtin_bit_cast(v16bf, w1v[(nt * NKB1 + kb) * 32 + lane]);
      acc = __builtin_amdgcn_wmma_f32_16x16x32_bf16(false, a, false, bm,
                                                    (short)0, acc, false, false);
    }
    // Store D-tile into A-swizzled sH1 (col is GEMM2's K index).
    const int kb2 = col >> 5;
    const int kk  = col & 31;
    const int ahi = (kk >> 3) & 1;
    const int e   = (kk & 7) + (kk >= 16 ? 8 : 0);
#pragma unroll
    for (int r = 0; r < 8; ++r) {
      const int m = r + hi * 8;            // row&15 within this wave's M-tile
      sH1[(((wave * NKB2 + kb2) * 32) + m + ahi * 16) * 16 + e] = (bf16)lrelu(acc[r]);
    }
  }
  __syncthreads();

  // GEMM2: H2[64][64] = lrelu(H1[64][128pad] @ W2[128][64] + b2)
  const u32x8* w2v = (const u32x8*)w2sw;
  for (int nt = 0; nt < NNT2; ++nt) {
    const int col = nt * 16 + nlo;
    const float bias = b2[col];
    v8f acc;
#pragma unroll
    for (int r = 0; r < 8; ++r) acc[r] = bias;
#pragma unroll
    for (int kb = 0; kb < NKB2; ++kb) {
      v16bf a  = __builtin_bit_cast(v16bf, sH1v[(wave * NKB2 + kb) * 32 + lane]);
      v16bf bm = __builtin_bit_cast(v16bf, w2v[(nt * NKB2 + kb) * 32 + lane]);
      acc = __builtin_amdgcn_wmma_f32_16x16x32_bf16(false, a, false, bm,
                                                    (short)0, acc, false, false);
    }
#pragma unroll
    for (int r = 0; r < 8; ++r)
      sH2[rbase + r + hi * 8][col] = lrelu(acc[r]);
  }
  __syncthreads();

  if (tid < DOUT) {
    float m = NEG_INF;
    for (int r = 0; r < KDN; ++r) m = fmaxf(m, sH2[r][tid]);
    dnout[((size_t)b * NPTS + n) * DOUT + tid] = m;
  }
}

// ---------------------------------------------------------------------------
// Sparse path (32 nodes, k=2) — tiny; plain VALU kernels.
// ---------------------------------------------------------------------------
__global__ __launch_bounds__(32) void sp_knn_edge(const float* __restrict__ spxt,
                                                  float* __restrict__ spE) {
  const int b = blockIdx.x;
  const int i = threadIdx.x;                  // 0..31
  const float* X = spxt + (size_t)b * NSTK * CFEA;
  const float* xi = X + i * CFEA;
  float x2i = 0.f;
  for (int c = 0; c < CFEA; ++c) x2i += xi[c] * xi[c];
  __shared__ float sx2[NSTK];
  sx2[i] = x2i;
  __syncthreads();

  float bv = NEG_INF, sv = NEG_INF;
  int bi = 0, si = 0;
  for (int j = 0; j < NSTK; ++j) {
    const float* xj = X + j * CFEA;
    float dot = 0.f;
    for (int c = 0; c < CFEA; ++c) dot += xi[c] * xj[c];
    const float v = 2.f * dot - x2i - sx2[j];
    if (v > bv)      { sv = bv; si = bi; bv = v; bi = j; }
    else if (v > sv) { sv = v; si = j; }
  }
  float* E = spE + ((size_t)b * (NSTK * KSP) + i * KSP) * TWO_C;
  const float* x0 = X + bi * CFEA;
  const float* x1 = X + si * CFEA;
  for (int c = 0; c < CFEA; ++c) {
    E[c]                 = x0[c] - xi[c];
    E[CFEA + c]          = xi[c];
    E[TWO_C + c]         = x1[c] - xi[c];
    E[TWO_C + CFEA + c]  = xi[c];
  }
}

__global__ __launch_bounds__(256) void sp_gemm_lrelu(const float* __restrict__ X,
                                                     const float* __restrict__ W,
                                                     const float* __restrict__ bias,
                                                     float* __restrict__ Y,
                                                     int R, int Cin, int O) {
  const int t = blockIdx.x * 256 + threadIdx.x;
  if (t >= R * O) return;
  const int r = t / O, o = t % O;
  const float* x = X + (size_t)r * Cin;
  const float* w = W + (size_t)o * Cin;
  float s = bias[o];
  for (int c = 0; c < Cin; ++c) s += x[c] * w[c];
  Y[t] = lrelu(s);
}

__global__ __launch_bounds__(256) void sp_max_out(const float* __restrict__ H2,
                                                  float* __restrict__ out) {
  const int t = blockIdx.x * 256 + threadIdx.x;   // [b][c][s] flat, 16384
  if (t >= BS * SP_OUT * NSTK) return;
  const int b = t / (SP_OUT * NSTK);
  const int rem = t % (SP_OUT * NSTK);
  const int c = rem / NSTK, s = rem % NSTK;
  const float a0 = H2[((size_t)b * (NSTK * KSP) + s * KSP + 0) * SP_OUT + c];
  const float a1 = H2[((size_t)b * (NSTK * KSP) + s * KSP + 1) * SP_OUT + c];
  out[t] = fmaxf(a0, a1);
}

// ---------------------------------------------------------------------------
// DownSample: Conv2d kernel (1,3), stride (1,2), pad (0,1) + LeakyReLU.
// dnout layout [b][n=s*256+p][64]; out [b][64][32][128].
// ---------------------------------------------------------------------------
__global__ __launch_bounds__(128) void down_sample(const float* __restrict__ dnout,
                                                   const float* __restrict__ W,
                                                   const float* __restrict__ bias,
                                                   float* __restrict__ out) {
  const int blk = blockIdx.x;                 // b*NSTK + s
  const int b = blk / NSTK, s = blk % NSTK;
  const int q = threadIdx.x;                  // 0..127
  const float* xb = dnout + ((size_t)b * NPTS + s * NPNT) * DOUT;
  for (int o = 0; o < DOUT; ++o) {
    float acc = bias[o];
    const float* w = W + (size_t)o * DOUT * 3;
#pragma unroll
    for (int t = 0; t < 3; ++t) {
      const int h = 2 * q + t - 1;
      if (h < 0 || h >= NPNT) continue;
      const float* x = xb + (size_t)h * DOUT;
      for (int i = 0; i < DOUT; ++i) acc += w[i * 3 + t] * x[i];
    }
    out[(((size_t)b * DOUT + o) * NSTK + s) * 128 + q] = lrelu(acc);
  }
}

// ---------------------------------------------------------------------------
// Launcher
// ---------------------------------------------------------------------------
extern "C" void kernel_launch(void* const* d_in, const int* in_sizes, int n_in,
                              void* d_out, int out_size, void* d_ws, size_t ws_size,
                              hipStream_t stream) {
  const float* sparse_fea = (const float*)d_in[0];
  const float* dense_fea  = (const float*)d_in[1];
  const float* sp_W1 = (const float*)d_in[2];
  const float* sp_b1 = (const float*)d_in[3];
  const float* sp_W2 = (const float*)d_in[4];
  const float* sp_b2 = (const float*)d_in[5];
  const float* dn_W1 = (const float*)d_in[6];
  const float* dn_b1 = (const float*)d_in[7];
  const float* dn_W2 = (const float*)d_in[8];
  const float* dn_b2 = (const float*)d_in[9];
  const float* ds_W  = (const float*)d_in[10];
  const float* ds_b  = (const float*)d_in[11];
  float* out = (float*)d_out;

  // Workspace carve-up (256B aligned)
  char* w = (char*)d_ws;
  auto carve = [&](size_t bytes) -> void* {
    void* p = (void*)w;
    w += (bytes + 255) & ~(size_t)255;
    return p;
  };
  float* xtf   = (float*)carve((size_t)BS * NPTS * CFEA * 4);        // 12.6 MB
  float* x2    = (float*)carve((size_t)BS * NPTS * 4);               // 128 KB
  int*   knn   = (int*)  carve((size_t)BS * NPTS * KDN * 4);         // 6.5 MB
  bf16*  w1sw  = (bf16*) carve((size_t)NNT1 * NKB1 * 512 * 2);       // 42 KB
  bf16*  w2sw  = (bf16*) carve((size_t)NNT2 * NKB2 * 512 * 2);       // 16 KB
  float* dnout = (float*)carve((size_t)BS * NPTS * DOUT * 4);        // 8.4 MB
  float* spxt  = (float*)carve((size_t)BS * NSTK * CFEA * 4);
  float* spE   = (float*)carve((size_t)BS * NSTK * KSP * TWO_C * 4);
  float* spH1  = (float*)carve((size_t)BS * NSTK * KSP * MID_SP * 4);
  float* spH2  = (float*)carve((size_t)BS * NSTK * KSP * SP_OUT * 4);

  // Stage 1: features + weight conversion
  build_xt<<<BS * NSTK, 256, 0, stream>>>(sparse_fea, dense_fea, xtf, x2);
  build_sp_xt<<<BS, 96, 0, stream>>>(sparse_fea, dense_fea, spxt);
  conv_weights<<<(NNT1 * NKB1 * 512 + 255) / 256, 256, 0, stream>>>(dn_W1, dn_W2, w1sw, w2sw);

  // Stage 2: dense KNN + WMMA edge-MLP + max over neighbors
  knn_topk<<<BS * (NPTS / TI), 256, 0, stream>>>(xtf, x2, knn);
  dense_edge_gemm<<<BS * NPTS, 128, 0, stream>>>(xtf, knn, w1sw, dn_b1, w2sw, dn_b2, dnout);

  // Stage 3: sparse path -> d_out[0 .. 16384)
  sp_knn_edge<<<BS, 32, 0, stream>>>(spxt, spE);
  {
    const int R = BS * NSTK * KSP;   // 256 edge rows total
    sp_gemm_lrelu<<<(R * MID_SP + 255) / 256, 256, 0, stream>>>(spE, sp_W1, sp_b1, spH1, R, TWO_C, MID_SP);
    sp_gemm_lrelu<<<(R * SP_OUT + 255) / 256, 256, 0, stream>>>(spH1, sp_W2, sp_b2, spH2, R, MID_SP, SP_OUT);
  }
  sp_max_out<<<(BS * SP_OUT * NSTK + 255) / 256, 256, 0, stream>>>(spH2, out);

  // Stage 4: down-sample conv -> d_out[16384 .. ]
  down_sample<<<BS * NSTK, 128, 0, stream>>>(dnout, ds_W, ds_b, out + BS * SP_OUT * NSTK);
}
